// MultiHeadSelfAttention_82927228552104
// MI455X (gfx1250) — matrix operand order repaired and checked by reading, not compiled, not run on hardware
//
#include <hip/hip_runtime.h>
#include <hip/hip_bf16.h>
#include <stdint.h>

// ---------------------------------------------------------------------------
// Problem constants (from reference): B=4, S=2048, E=1024, H=16, Dh=64
// ---------------------------------------------------------------------------
namespace cfg {
constexpr int kB  = 4;
constexpr int kS  = 2048;
constexpr int kE  = 1024;
constexpr int kH  = 16;
constexpr int kDh = 64;
constexpr int kM  = kB * kS;          // 8192 rows of the activation matrix
constexpr int kSROW = kS + 1;         // padded LDS row stride (bank-conflict free)
}

typedef __bf16 bf16;
typedef __attribute__((ext_vector_type(16))) __bf16 v16bf;
typedef __attribute__((ext_vector_type(8)))  float  v8f;

union ABFrag {
    v16bf    v;
    uint32_t u[8];
    bf16     h[16];
};

__device__ __forceinline__ v8f wmma_bf16(const ABFrag& a, const ABFrag& b, v8f c) {
    // v_wmma_f32_16x16x32_bf16: D = A(16x32) * B(32x16) + C(16x16 f32)
    return __builtin_amdgcn_wmma_f32_16x16x32_bf16(
        /*neg_a=*/false, a.v, /*neg_b=*/false, b.v,
        /*c_mod=*/(short)0, c, /*reuse_a=*/false, /*reuse_b=*/false);
}

// A fragment (16x32 bf16, row-major source, row stride ld elements).
// ISA layout: lane m = lane&15, K-group = lane>>4; VGPR r holds packed K pair:
//   K = (r>=4 ? 16 : 0) + (r&3)*2 + (lane>>4)*8  (+1 in high half)
__device__ __forceinline__ ABFrag load_a_frag(const bf16* base, int ld) {
    const int lane = threadIdx.x & 31;
    const int m = lane & 15, kg = lane >> 4;
    ABFrag f;
#pragma unroll
    for (int r = 0; r < 8; ++r) {
        const int k = ((r & 4) << 2) + ((r & 3) << 1) + (kg << 3);
        f.u[r] = *(const uint32_t*)(base + (size_t)m * ld + k);
    }
    return f;
}

// B fragment (32x16 bf16) from an N-major source: element(k,n) = baseT[n*ld + k].
// ISA layout: same K order as A: K = (r>=4 ? 16 : 0) + (r&3)*2 + (lane>>4)*8 (+1 in high half).
__device__ __forceinline__ ABFrag load_b_frag(const bf16* baseT, int ld) {
    const int lane = threadIdx.x & 31;
    const int n = lane & 15, kg = lane >> 4;
    ABFrag f;
#pragma unroll
    for (int r = 0; r < 8; ++r)
        f.u[r] = *(const uint32_t*)(baseT + (size_t)n * ld + ((r & 4) << 2) + ((r & 3) << 1) + (kg << 3));
    return f;
}

// ---------------------------------------------------------------------------
// Conversion kernels
// ---------------------------------------------------------------------------
__global__ void cvt_f32_to_bf16(const float* __restrict__ in,
                                bf16* __restrict__ out, int n) {
    int i = blockIdx.x * 256 + threadIdx.x;
    if (i < n) out[i] = (bf16)in[i];
}

// Wt[n*E + k] = (bf16) W[k*E + n]   (transpose so B-fragments load contiguous K)
__global__ void cvt_transpose_w(const float* __restrict__ W,
                                bf16* __restrict__ Wt) {
    int i = blockIdx.x * 256 + threadIdx.x;
    if (i >= cfg::kE * cfg::kE) return;
    int n = i / cfg::kE, k = i - n * cfg::kE;
    Wt[i] = (bf16)W[(size_t)k * cfg::kE + n];
}

// ---------------------------------------------------------------------------
// WMMA GEMM: C[M,N] = A(bf16,[M,K] row-major) * Bt(bf16,[N,K] n-major) + bias
//   mode 0: store bf16 row-major             (Q, K projections)
//   mode 1: store bf16 transposed [B,H,Dh,S] (V projection)
//   mode 2: store fp32 row-major, += resid   (O projection + residual)
// Block = 256 threads = 8 waves tiling 64x128 of C; each wave register-blocks
// a 32x32 tile (2x2 WMMA tiles) so every A/B fragment feeds two WMMAs.
// ---------------------------------------------------------------------------
__device__ __forceinline__ void gemm_epilogue(v8f c, int tile_m, int tile_n, int N,
                                              int mode, const float* bias,
                                              const float* resid_in, void* out) {
    const int lane = threadIdx.x & 31;
    const int m0 = tile_m + 8 * (lane >> 4);
    const int n  = tile_n + (lane & 15);
    const float bi = bias[n];
    if (mode == 0) {
        bf16* O = (bf16*)out;
#pragma unroll
        for (int r = 0; r < 8; ++r)
            O[(size_t)(m0 + r) * N + n] = (bf16)(c[r] + bi);
    } else if (mode == 1) {
        bf16* O = (bf16*)out;                       // [B, H, Dh, S]
        const int h = n >> 6, dh = n & 63;
#pragma unroll
        for (int r = 0; r < 8; ++r) {
            const int row = m0 + r;
            const int bb = row >> 11, s = row & (cfg::kS - 1);
            O[(((size_t)bb * cfg::kH + h) * cfg::kDh + dh) * cfg::kS + s] =
                (bf16)(c[r] + bi);
        }
    } else {
        float* O = (float*)out;
#pragma unroll
        for (int r = 0; r < 8; ++r) {
            const size_t idx = (size_t)(m0 + r) * N + n;
            O[idx] = c[r] + bi + resid_in[idx];
        }
    }
}

__global__ __launch_bounds__(256)
void gemm_wmma(const bf16* __restrict__ A, const bf16* __restrict__ Bt,
               const float* __restrict__ bias, const float* __restrict__ resid_in,
               void* __restrict__ out, int M, int N, int K, int mode) {
    const int wave = threadIdx.x >> 5;
    const int wr = wave >> 2;                      // 0..1  (row group of 32)
    const int wc = wave & 3;                       // 0..3  (col group of 32)
    const int tm = blockIdx.y * 64 + wr * 32;      // rows [tm, tm+32)
    const int tn = blockIdx.x * 128 + wc * 32;     // cols [tn, tn+32)
    if (tm >= M || tn >= N) return;

    const bf16* A0 = A  + (size_t)tm * K;
    const bf16* A1 = A  + (size_t)(tm + 16) * K;
    const bf16* B0 = Bt + (size_t)tn * K;
    const bf16* B1 = Bt + (size_t)(tn + 16) * K;

    v8f c00 = {}, c01 = {}, c10 = {}, c11 = {};
#pragma unroll 4
    for (int k0 = 0; k0 < K; k0 += 32) {
        ABFrag a0 = load_a_frag(A0 + k0, K);
        ABFrag a1 = load_a_frag(A1 + k0, K);
        ABFrag b0 = load_b_frag(B0 + k0, K);
        ABFrag b1 = load_b_frag(B1 + k0, K);
        c00 = wmma_bf16(a0, b0, c00);
        c01 = wmma_bf16(a0, b1, c01);
        c10 = wmma_bf16(a1, b0, c10);
        c11 = wmma_bf16(a1, b1, c11);
    }

    gemm_epilogue(c00, tm,      tn,      N, mode, bias, resid_in, out);
    gemm_epilogue(c01, tm,      tn + 16, N, mode, bias, resid_in, out);
    gemm_epilogue(c10, tm + 16, tn,      N, mode, bias, resid_in, out);
    gemm_epilogue(c11, tm + 16, tn + 16, N, mode, bias, resid_in, out);
}

// ---------------------------------------------------------------------------
// Attention: one block = 8 waves handles 16 query rows of one (b,h).
//  Phase 1: waves compute masked score tiles (2 WMMA each) into LDS stripe.
//  Phase 2: per-row max / sum-of-exp reduction.
//  Phase 3: write softmax weights to d_out, overwrite LDS with probs.
//  Phase 4: P*V via WMMA (4 output d-tiles per wave), LDS partial reduce.
// ---------------------------------------------------------------------------
__global__ __launch_bounds__(256)
void attn_wmma(const bf16* __restrict__ Q, const bf16* __restrict__ Kb,
               const bf16* __restrict__ Vt, bf16* __restrict__ attn_out,
               float* __restrict__ w_out) {
    using namespace cfg;
    extern __shared__ float smem[];
    float* s_sc   = smem;                       // [16][kSROW] scores -> probs
    float* s_part = smem + 16 * kSROW;          // [8][16][64] PV partials
    float* s_max  = s_part + 8 * 16 * 64;       // [16]
    float* s_sum  = s_max + 16;                 // [16]

    const int wave = threadIdx.x >> 5;
    const int lane = threadIdx.x & 31;
    const int m_l  = lane & 15;
    const int kg   = lane >> 4;

    const int qt = blockIdx.x, h = blockIdx.y, b = blockIdx.z;
    const int qbase = qt * 16;
    const int nkt   = qt + 1;        // causal: 16-wide key tiles covered
    const int ncov  = nkt * 16;

    // Q fragments (16 rows x 64 dims = two 16x32 A frags), reused all iters.
    const bf16* Qrow = Q + ((size_t)(b * kS + qbase) * kE) + h * kDh;
    const ABFrag aq0 = load_a_frag(Qrow, kE);
    const ABFrag aq1 = load_a_frag(Qrow + 32, kE);

    // ---- Phase 1: scores = QK^T / sqrt(Dh), causal mask, into LDS ----
    for (int kt = wave; kt < nkt; kt += 8) {
        const bf16* Kt = Kb + ((size_t)(b * kS + kt * 16) * kE) + h * kDh;
        ABFrag bk0 = load_b_frag(Kt, kE);        // B(k=d, n=key) contiguous d
        ABFrag bk1 = load_b_frag(Kt + 32, kE);
        v8f c = {};
        c = wmma_bf16(aq0, bk0, c);
        c = wmma_bf16(aq1, bk1, c);
        const int n = kt * 16 + m_l;             // global key index
#pragma unroll
        for (int r = 0; r < 8; ++r) {
            const int m = r + 8 * kg;            // local query row
            float s = c[r] * 0.125f;             // 1/sqrt(64)
            if (n > qbase + m) s = -1e9f;        // causal mask
            s_sc[m * kSROW + n] = s;
        }
    }
    __syncthreads();

    // ---- Phase 2: per-row max and sum(exp) (wave w handles rows 2w,2w+1) ----
    for (int rr = 0; rr < 2; ++rr) {
        const int m = wave * 2 + rr;
        float mx = -1e30f;
        for (int j = lane; j < ncov; j += 32)
            mx = fmaxf(mx, s_sc[m * kSROW + j]);
#pragma unroll
        for (int o = 16; o; o >>= 1) mx = fmaxf(mx, __shfl_xor(mx, o, 32));
        float sm = 0.f;
        for (int j = lane; j < ncov; j += 32)
            sm += __expf(s_sc[m * kSROW + j] - mx);
#pragma unroll
        for (int o = 16; o; o >>= 1) sm += __shfl_xor(sm, o, 32);
        if (lane == 0) { s_max[m] = mx; s_sum[m] = sm; }
    }
    __syncthreads();

    // ---- Phase 3: write weights to global, LDS scores -> probs ----
    float* wbase = w_out + ((size_t)(b * kH + h) * kS + qbase) * kS;
    for (int idx = threadIdx.x; idx < 16 * kS; idx += 256) {
        const int m = idx >> 11, k = idx & (kS - 1);
        float p = 0.f;
        if (k < ncov) {
            p = __expf(s_sc[m * kSROW + k] - s_max[m]) / s_sum[m];
            s_sc[m * kSROW + k] = p;
        }
        wbase[(size_t)m * kS + k] = p;           // zeros beyond causal bound
    }
    __syncthreads();

    // ---- Phase 4: O = P * V  (waves split the key dimension) ----
    v8f o0 = {}, o1 = {}, o2 = {}, o3 = {};
    const bf16* Vhead = Vt + (size_t)(b * kH + h) * kDh * kS;   // [Dh][S]
    for (int kc = wave * 32; kc < ncov; kc += 256) {
        ABFrag ap;                                // probs 16x32 as A fragment
#pragma unroll
        for (int r = 0; r < 8; ++r) {
            const int k = kc + ((r & 4) << 2) + ((r & 3) << 1) + (kg << 3);
            const float p0 = (k     < ncov) ? s_sc[m_l * kSROW + k]     : 0.f;
            const float p1 = (k + 1 < ncov) ? s_sc[m_l * kSROW + k + 1] : 0.f;
            ap.h[2 * r]     = (bf16)p0;
            ap.h[2 * r + 1] = (bf16)p1;
        }
        const bf16* Vb = Vhead + kc;              // B(k=key, n=d) contiguous key
        ABFrag b0 = load_b_frag(Vb,            kS);
        ABFrag b1 = load_b_frag(Vb + 16 * kS,  kS);
        ABFrag b2 = load_b_frag(Vb + 32 * kS,  kS);
        ABFrag b3 = load_b_frag(Vb + 48 * kS,  kS);
        o0 = wmma_bf16(ap, b0, o0);
        o1 = wmma_bf16(ap, b1, o1);
        o2 = wmma_bf16(ap, b2, o2);
        o3 = wmma_bf16(ap, b3, o3);
    }

    float* pb = s_part + wave * 16 * 64;
#pragma unroll
    for (int r = 0; r < 8; ++r) {
        const int m = r + 8 * kg;
        pb[m * 64 +  0 + m_l] = o0[r];
        pb[m * 64 + 16 + m_l] = o1[r];
        pb[m * 64 + 32 + m_l] = o2[r];
        pb[m * 64 + 48 + m_l] = o3[r];
    }
    __syncthreads();

    for (int idx = threadIdx.x; idx < 16 * 64; idx += 256) {
        const int m = idx >> 6, d = idx & 63;
        float acc = 0.f;
#pragma unroll
        for (int wv = 0; wv < 8; ++wv) acc += s_part[wv * 1024 + idx];
        attn_out[((size_t)(b * kS + qbase + m) * kE) + h * kDh + d] = (bf16)acc;
    }
}

// ---------------------------------------------------------------------------
// LayerNorm: one block (256 threads) per row of 1024.
// ---------------------------------------------------------------------------
__global__ __launch_bounds__(256)
void layernorm_rows(const float* __restrict__ resid,
                    const float* __restrict__ gamma,
                    const float* __restrict__ beta,
                    float* __restrict__ out) {
    const int row = blockIdx.x;
    const float* r = resid + (size_t)row * cfg::kE;
    float v[4], s = 0.f, s2 = 0.f;
#pragma unroll
    for (int i = 0; i < 4; ++i) {
        v[i] = r[threadIdx.x + 256 * i];
        s += v[i]; s2 += v[i] * v[i];
    }
#pragma unroll
    for (int o = 16; o; o >>= 1) {
        s  += __shfl_xor(s,  o, 32);
        s2 += __shfl_xor(s2, o, 32);
    }
    __shared__ float ss[8], ss2[8];
    if ((threadIdx.x & 31) == 0) { ss[threadIdx.x >> 5] = s; ss2[threadIdx.x >> 5] = s2; }
    __syncthreads();
    float ts = 0.f, ts2 = 0.f;
#pragma unroll
    for (int wv = 0; wv < 8; ++wv) { ts += ss[wv]; ts2 += ss2[wv]; }
    const float mu   = ts  * (1.f / cfg::kE);
    const float var  = ts2 * (1.f / cfg::kE) - mu * mu;
    const float rstd = rsqrtf(var + 1e-6f);
#pragma unroll
    for (int i = 0; i < 4; ++i) {
        const int j = threadIdx.x + 256 * i;
        out[(size_t)row * cfg::kE + j] = (v[i] - mu) * rstd * gamma[j] + beta[j];
    }
}

// ---------------------------------------------------------------------------
// Launch: 11 inputs in reference order; d_out = [normed (8.4M f32) | weights].
// Workspace usage ~120 MB: Xb | Wq/Wk/Wv/Wo (bf16,T) | Qb | Kb | Vt | Attn | resid
// ---------------------------------------------------------------------------
extern "C" void kernel_launch(void* const* d_in, const int* in_sizes, int n_in,
                              void* d_out, int out_size, void* d_ws, size_t ws_size,
                              hipStream_t stream) {
    (void)in_sizes; (void)n_in; (void)out_size; (void)ws_size;
    using namespace cfg;

    const float* X     = (const float*)d_in[0];
    const float* Wq    = (const float*)d_in[1];
    const float* bq    = (const float*)d_in[2];
    const float* Wk    = (const float*)d_in[3];
    const float* bk    = (const float*)d_in[4];
    const float* Wv    = (const float*)d_in[5];
    const float* bv    = (const float*)d_in[6];
    const float* Wo    = (const float*)d_in[7];
    const float* bo    = (const float*)d_in[8];
    const float* gamma = (const float*)d_in[9];
    const float* beta  = (const float*)d_in[10];

    char* ws = (char*)d_ws;
    bf16* Xb   = (bf16*)ws;  ws += (size_t)kM * kE * sizeof(bf16);
    bf16* Wqt  = (bf16*)ws;  ws += (size_t)kE * kE * sizeof(bf16);
    bf16* Wkt  = (bf16*)ws;  ws += (size_t)kE * kE * sizeof(bf16);
    bf16* Wvt  = (bf16*)ws;  ws += (size_t)kE * kE * sizeof(bf16);
    bf16* Wot  = (bf16*)ws;  ws += (size_t)kE * kE * sizeof(bf16);
    bf16* Qb   = (bf16*)ws;  ws += (size_t)kM * kE * sizeof(bf16);
    bf16* Kb   = (bf16*)ws;  ws += (size_t)kM * kE * sizeof(bf16);
    bf16* Vt   = (bf16*)ws;  ws += (size_t)kM * kE * sizeof(bf16);
    bf16* Attn = (bf16*)ws;  ws += (size_t)kM * kE * sizeof(bf16);
    float* resid = (float*)ws;

    float* out_norm = (float*)d_out;
    float* out_w    = (float*)d_out + (size_t)kM * kE;

    // 1) precision conversion / weight transposes
    cvt_f32_to_bf16<<<(kM * kE + 255) / 256, 256, 0, stream>>>(X, Xb, kM * kE);
    const int wgrid = (kE * kE + 255) / 256;
    cvt_transpose_w<<<wgrid, 256, 0, stream>>>(Wq, Wqt);
    cvt_transpose_w<<<wgrid, 256, 0, stream>>>(Wk, Wkt);
    cvt_transpose_w<<<wgrid, 256, 0, stream>>>(Wv, Wvt);
    cvt_transpose_w<<<wgrid, 256, 0, stream>>>(Wo, Wot);

    // 2) Q/K/V projections (WMMA bf16, fp32 accum); block tiles 64x128
    dim3 ggrid(kE / 128, kM / 64);               // (8, 128)
    gemm_wmma<<<ggrid, 256, 0, stream>>>(Xb, Wqt, bq, nullptr, Qb, kM, kE, kE, 0);
    gemm_wmma<<<ggrid, 256, 0, stream>>>(Xb, Wkt, bk, nullptr, Kb, kM, kE, kE, 0);
    gemm_wmma<<<ggrid, 256, 0, stream>>>(Xb, Wvt, bv, nullptr, Vt, kM, kE, kE, 1);

    // 3) attention (scores + softmax + weights-out + P*V)
    const size_t smem_bytes = (size_t)(16 * kSROW + 8 * 16 * 64 + 32) * sizeof(float);
    attn_wmma<<<dim3(kS / 16, kH, kB), 256, smem_bytes, stream>>>(
        Qb, Kb, Vt, Attn, out_w);

    // 4) output projection + residual, then layernorm
    gemm_wmma<<<ggrid, 256, 0, stream>>>(Attn, Wot, bo, X, resid, kM, kE, kE, 2);
    layernorm_rows<<<kM, 256, 0, stream>>>(resid, gamma, beta, out_norm);
}
